// AutoCorrelation_83485574300200
// MI455X (gfx1250) — matrix operand order repaired and checked by reading, not compiled, not run on hardware
//
#include <hip/hip_runtime.h>
#include <hip/hip_bf16.h>
#include <stdint.h>
#include <math.h>

#define B_    8
#define T_    2048
#define C_    1024
#define TOPK_ 38

typedef __attribute__((ext_vector_type(16))) __bf16 v16bf;
typedef __attribute__((ext_vector_type(8)))  float  v8f;
typedef __attribute__((ext_vector_type(4)))  int    v4i_t;

union Frag { v16bf v; uint32_t u[8]; };

#if defined(__HIP_DEVICE_COMPILE__) && __has_builtin(__builtin_amdgcn_global_load_async_to_lds_b128)
#define HAVE_ASYNC 1
#else
#define HAVE_ASYNC 0
#endif

#if HAVE_ASYNC
#if __has_builtin(__builtin_amdgcn_s_wait_asynccnt)
#define WAIT_ASYNC(n) __builtin_amdgcn_s_wait_asynccnt(n)
#else
#define WAIT_ASYNC(n) asm volatile("s_wait_asynccnt %0" ::"i"(n) : "memory")
#endif
#else
#define WAIT_ASYNC(n) ((void)0)
#endif

__device__ __forceinline__ uint16_t f2bf(float f) {
  uint32_t x = __float_as_uint(f);
  uint32_t r = x + 0x7FFFu + ((x >> 16) & 1u);   // round-to-nearest-even
  return (uint16_t)(r >> 16);
}
__device__ __forceinline__ float bf2f(uint16_t h) {
  return __uint_as_float(((uint32_t)h) << 16);
}

// copy 8 bf16 (16B) global -> LDS; async path uses GLOBAL_LOAD_ASYNC_TO_LDS_B128
__device__ __forceinline__ void cp16_async(const uint16_t* g, uint16_t* l) {
#if HAVE_ASYNC
  typedef __attribute__((address_space(1))) v4i_t glb_v4i;
  typedef __attribute__((address_space(3))) v4i_t lds_v4i;
  __builtin_amdgcn_global_load_async_to_lds_b128(
      (glb_v4i*)(uintptr_t)g, (lds_v4i*)(uintptr_t)l, 0, 0);
#else
  *(uint4*)l = *(const uint4*)g;
#endif
}

// ---------------- conversion / init ----------------
__global__ void k_f2bf(const float* __restrict__ src, uint16_t* __restrict__ dst, int n) {
  int i = blockIdx.x * blockDim.x + threadIdx.x;
  int stride = gridDim.x * blockDim.x;
  for (; i < n; i += stride) dst[i] = f2bf(src[i]);
}
// transpose-convert a 1024x1024 weight: dst[n*1024 + k] = bf16(src[k*1024 + n])
__global__ void k_f2bfT(const float* __restrict__ src, uint16_t* __restrict__ dst) {
  int i = blockIdx.x * blockDim.x + threadIdx.x;
  int stride = gridDim.x * blockDim.x;
  for (; i < C_ * C_; i += stride) {
    int k = i >> 10, n = i & 1023;
    dst[n * C_ + k] = f2bf(src[i]);
  }
}
__global__ void k_zero(float* __restrict__ p, int n) {
  int i = blockIdx.x * blockDim.x + threadIdx.x;
  int stride = gridDim.x * blockDim.x;
  for (; i < n; i += stride) p[i] = 0.0f;
}

// ---------------- bf16 WMMA GEMM: out = A[MxK] * Bt[NxK]^T + bias ----------------
// 256 threads = 8 waves (4 along M x 2 along N); block tile 256x128; wave tile 64x64
// double-buffered LDS, async global->LDS staging, 16 WMMA per K-step of 32
template <bool OUT_BF16>
__global__ __launch_bounds__(256)
void k_gemm(const uint16_t* __restrict__ A, const uint16_t* __restrict__ Bt,
            const float* __restrict__ bias, void* __restrict__ outp,
            int M, int N, int K)
{
  __shared__ __attribute__((aligned(16))) uint16_t As[2][256][32];
  __shared__ __attribute__((aligned(16))) uint16_t Bs[2][128][32];

  const int tid  = threadIdx.x;
  const int lane = tid & 31, wave = tid >> 5;
  const int wm = wave & 3, wn = wave >> 2;       // 4 x 2 wave grid
  const int half = lane >> 4, qn = lane & 15;
  const int m0 = blockIdx.y * 256, n0 = blockIdx.x * 128;

  v8f zero = {0.f,0.f,0.f,0.f,0.f,0.f,0.f,0.f};
  v8f acc[4][4];
  #pragma unroll
  for (int mt = 0; mt < 4; ++mt)
    #pragma unroll
    for (int nt = 0; nt < 4; ++nt) acc[mt][nt] = zero;

  // stage one K-slab (256x32 A, 128x32 Bt) into buffer `buf`
  auto stage = [&](int buf, int k0) {
    #pragma unroll
    for (int i = 0; i < 4; ++i) {               // A: 1024 16B-chunks
      int ch = (i << 8) + tid;
      int r = ch >> 2, sg = (ch & 3) << 3;
      cp16_async(A + (size_t)(m0 + r) * K + k0 + sg, &As[buf][r][sg]);
    }
    #pragma unroll
    for (int i = 0; i < 2; ++i) {               // B: 512 16B-chunks
      int ch = (i << 8) + tid;
      int r = ch >> 2, sg = (ch & 3) << 3;
      cp16_async(Bt + (size_t)(n0 + r) * K + k0 + sg, &Bs[buf][r][sg]);
    }
  };

  stage(0, 0);
  for (int k0 = 0; k0 < K; k0 += 32) {
    const int cur = (k0 >> 5) & 1;
    if (k0 + 32 < K) { stage(cur ^ 1, k0 + 32); WAIT_ASYNC(6); }
    else             { WAIT_ASYNC(0); }
    __syncthreads();

    Frag af[4];
    #pragma unroll
    for (int mt = 0; mt < 4; ++mt) {
      int row = wm * 64 + mt * 16 + qn;
      #pragma unroll
      for (int r = 0; r < 8; ++r) {
        int kb = (r < 4 ? 2 * r : 8 + 2 * r) + 8 * half;   // ISA A 16x32 layout
        af[mt].u[r] = *(const uint32_t*)&As[cur][row][kb];
      }
    }
    Frag bfg[4];
    #pragma unroll
    for (int nt = 0; nt < 4; ++nt) {
      int col = wn * 64 + nt * 16 + qn;
      #pragma unroll
      for (int r = 0; r < 8; ++r) {
        int kb = (r < 4 ? 2 * r : 8 + 2 * r) + 8 * half;   // ISA B 32x16 layout
        bfg[nt].u[r] = *(const uint32_t*)&Bs[cur][col][kb];
      }
    }
    #pragma unroll
    for (int mt = 0; mt < 4; ++mt)
      #pragma unroll
      for (int nt = 0; nt < 4; ++nt)
        acc[mt][nt] = __builtin_amdgcn_wmma_f32_16x16x32_bf16(
            false, af[mt].v, false, bfg[nt].v, (short)0, acc[mt][nt], false, false);

    __syncthreads();
  }

  #pragma unroll
  for (int nt = 0; nt < 4; ++nt) {
    int col = n0 + wn * 64 + nt * 16 + qn;
    float bv = bias[col];
    #pragma unroll
    for (int mt = 0; mt < 4; ++mt) {
      #pragma unroll
      for (int r = 0; r < 8; ++r) {
        int row = m0 + wm * 64 + mt * 16 + r + 8 * half;    // ISA C/D layout
        float v = acc[mt][nt][r] + bv;
        if constexpr (OUT_BF16) ((uint16_t*)outp)[(size_t)row * N + col] = f2bf(v);
        else                    ((float*)outp)[(size_t)row * N + col] = v;
      }
    }
  }
}

// ------- Gram tiles Q K^T, circular-diagonal reduction into mean[B,T] -------
// block tile 256(t) x 128(s), same WMMA micro-structure as k_gemm
__global__ __launch_bounds__(256)
void k_corr(const uint16_t* __restrict__ Qh, const uint16_t* __restrict__ Kh,
            float* __restrict__ meanv)
{
  __shared__ __attribute__((aligned(16))) uint16_t As[2][256][32];
  __shared__ __attribute__((aligned(16))) uint16_t Bs[2][128][32];
  __shared__ float diag[383];

  const int tid  = threadIdx.x;
  const int lane = tid & 31, wave = tid >> 5;
  const int wm = wave & 3, wn = wave >> 2;
  const int half = lane >> 4, qn = lane & 15;
  const int b = blockIdx.z;
  const int t0 = blockIdx.y * 256, s0 = blockIdx.x * 128;
  const uint16_t* Qb = Qh + (size_t)b * T_ * C_;
  const uint16_t* Kb = Kh + (size_t)b * T_ * C_;

  for (int i = tid; i < 383; i += 256) diag[i] = 0.0f;

  v8f zero = {0.f,0.f,0.f,0.f,0.f,0.f,0.f,0.f};
  v8f acc[4][4];
  #pragma unroll
  for (int mt = 0; mt < 4; ++mt)
    #pragma unroll
    for (int nt = 0; nt < 4; ++nt) acc[mt][nt] = zero;

  auto stage = [&](int buf, int k0) {
    #pragma unroll
    for (int i = 0; i < 4; ++i) {
      int ch = (i << 8) + tid;
      int r = ch >> 2, sg = (ch & 3) << 3;
      cp16_async(Qb + (size_t)(t0 + r) * C_ + k0 + sg, &As[buf][r][sg]);
    }
    #pragma unroll
    for (int i = 0; i < 2; ++i) {
      int ch = (i << 8) + tid;
      int r = ch >> 2, sg = (ch & 3) << 3;
      cp16_async(Kb + (size_t)(s0 + r) * C_ + k0 + sg, &Bs[buf][r][sg]);
    }
  };

  stage(0, 0);
  for (int k0 = 0; k0 < C_; k0 += 32) {
    const int cur = (k0 >> 5) & 1;
    if (k0 + 32 < C_) { stage(cur ^ 1, k0 + 32); WAIT_ASYNC(6); }
    else              { WAIT_ASYNC(0); }
    __syncthreads();

    Frag af[4];
    #pragma unroll
    for (int mt = 0; mt < 4; ++mt) {
      int row = wm * 64 + mt * 16 + qn;
      #pragma unroll
      for (int r = 0; r < 8; ++r) {
        int kb = (r < 4 ? 2 * r : 8 + 2 * r) + 8 * half;
        af[mt].u[r] = *(const uint32_t*)&As[cur][row][kb];
      }
    }
    Frag bfg[4];
    #pragma unroll
    for (int nt = 0; nt < 4; ++nt) {
      int col = wn * 64 + nt * 16 + qn;
      #pragma unroll
      for (int r = 0; r < 8; ++r) {
        int kb = (r < 4 ? 2 * r : 8 + 2 * r) + 8 * half;
        bfg[nt].u[r] = *(const uint32_t*)&Bs[cur][col][kb];
      }
    }
    #pragma unroll
    for (int mt = 0; mt < 4; ++mt)
      #pragma unroll
      for (int nt = 0; nt < 4; ++nt)
        acc[mt][nt] = __builtin_amdgcn_wmma_f32_16x16x32_bf16(
            false, af[mt].v, false, bfg[nt].v, (short)0, acc[mt][nt], false, false);

    __syncthreads();
  }

  // S[t,s] -> tau = (t - s) mod T; block-local diagonal sums in LDS, then global atomics
  #pragma unroll
  for (int mt = 0; mt < 4; ++mt)
    #pragma unroll
    for (int nt = 0; nt < 4; ++nt)
      #pragma unroll
      for (int r = 0; r < 8; ++r) {
        int i = wm * 64 + mt * 16 + r + 8 * half;   // 0..255
        int j = wn * 64 + nt * 16 + qn;             // 0..127
        atomicAdd(&diag[i - j + 127], acc[mt][nt][r]);
      }
  __syncthreads();
  for (int d = tid; d < 383; d += 256) {
    int tau = (t0 - s0 + d - 127) & (T_ - 1);
    atomicAdd(&meanv[b * T_ + tau], diag[d]);
  }
}

// ---------------- per-batch top-K + softmax ----------------
__global__ __launch_bounds__(256)
void k_topk(const float* __restrict__ meanv, float* __restrict__ topw, int* __restrict__ topd)
{
  const int b = blockIdx.x, tid = threadIdx.x;
  __shared__ float vals[T_];
  __shared__ float rv[256];
  __shared__ int   ri[256];
  __shared__ float selV[TOPK_];
  __shared__ int   selI[TOPK_];

  for (int i = tid; i < T_; i += 256) vals[i] = meanv[b * T_ + i] * (1.0f / (float)C_);
  __syncthreads();

  for (int k = 0; k < TOPK_; ++k) {
    float best = -1e30f; int bi = 0;
    for (int i = tid; i < T_; i += 256) { float v = vals[i]; if (v > best) { best = v; bi = i; } }
    rv[tid] = best; ri[tid] = bi;
    __syncthreads();
    for (int s = 128; s > 0; s >>= 1) {
      if (tid < s && rv[tid + s] > rv[tid]) { rv[tid] = rv[tid + s]; ri[tid] = ri[tid + s]; }
      __syncthreads();
    }
    if (tid == 0) { selV[k] = rv[0]; selI[k] = ri[0]; vals[ri[0]] = -1e30f; }
    __syncthreads();
  }

  if (tid == 0) {
    float mx = selV[0], s = 0.f, e[TOPK_];
    for (int k = 0; k < TOPK_; ++k) { e[k] = __expf(selV[k] - mx); s += e[k]; }
    float inv = 1.0f / s;
    for (int k = 0; k < TOPK_; ++k) {
      topw[b * TOPK_ + k] = e[k] * inv;
      topd[b * TOPK_ + k] = selI[k];
    }
  }
}

// -------- weighted circular gather with the reference's permute scramble --------
__global__ __launch_bounds__(256)
void k_agg(const uint16_t* __restrict__ Vh, const float* __restrict__ topw,
           const int* __restrict__ topd, uint16_t* __restrict__ aggH)
{
  __shared__ float ws[TOPK_];
  __shared__ int   ds[TOPK_];
  const int b = blockIdx.y, tid = threadIdx.x;
  if (tid < TOPK_) { ws[tid] = topw[b * TOPK_ + tid]; ds[tid] = topd[b * TOPK_ + tid]; }
  __syncthreads();

  int p = blockIdx.x * 256 + tid;          // index into T*C
  int t = p >> 10, c = p & 1023;
  int l  = ((t & 127) << 4) + (c >> 6);    // time index in values-space
  int ch = ((t >> 7) << 6) + (c & 63);     // channel h*E+e
  const uint16_t* Vb = Vh + (size_t)b * T_ * C_;

  float acc = 0.f;
  #pragma unroll 2
  for (int k = 0; k < TOPK_; ++k) {
    int src = (l + ds[k]) & (T_ - 1);
    acc += ws[k] * bf2f(Vb[(size_t)src * C_ + ch]);
  }
  aggH[((size_t)b * T_ + t) * C_ + c] = f2bf(acc);
}

// ---------------- host ----------------
extern "C" void kernel_launch(void* const* d_in, const int* in_sizes, int n_in,
                              void* d_out, int out_size, void* d_ws, size_t ws_size,
                              hipStream_t stream)
{
  (void)in_sizes; (void)n_in; (void)out_size; (void)ws_size;
  const float* x  = (const float*)d_in[0];
  const float* Wq = (const float*)d_in[1];
  const float* bq = (const float*)d_in[2];
  const float* Wk = (const float*)d_in[3];
  const float* bk = (const float*)d_in[4];
  const float* Wv = (const float*)d_in[5];
  const float* bv = (const float*)d_in[6];
  const float* Wp = (const float*)d_in[7];
  const float* bp = (const float*)d_in[8];
  float* out = (float*)d_out;

  const size_t BT  = (size_t)B_ * T_;      // 16384
  const size_t BTC = BT * C_;              // 16,777,216
  const size_t CC  = (size_t)C_ * C_;

  char* ws = (char*)d_ws;
  size_t off = 0;
  uint16_t* xh   = (uint16_t*)(ws + off); off += BTC * 2;   // reused as agg buffer
  uint16_t* wqh  = (uint16_t*)(ws + off); off += CC * 2;    // transposed weights (N x K)
  uint16_t* wkh  = (uint16_t*)(ws + off); off += CC * 2;
  uint16_t* wvh  = (uint16_t*)(ws + off); off += CC * 2;
  uint16_t* wph  = (uint16_t*)(ws + off); off += CC * 2;
  uint16_t* qh   = (uint16_t*)(ws + off); off += BTC * 2;
  uint16_t* kh   = (uint16_t*)(ws + off); off += BTC * 2;
  uint16_t* vh   = (uint16_t*)(ws + off); off += BTC * 2;
  float*    mnv  = (float*)(ws + off);    off += BT * 4;
  float*    topw = (float*)(ws + off);    off += 2048;
  int*      topd = (int*)(ws + off);      off += 2048;
  uint16_t* aggh = xh;

  k_f2bf <<<dim3(2048), dim3(256), 0, stream>>>(x,  xh, (int)BTC);
  k_f2bfT<<<dim3(512),  dim3(256), 0, stream>>>(Wq, wqh);
  k_f2bfT<<<dim3(512),  dim3(256), 0, stream>>>(Wk, wkh);
  k_f2bfT<<<dim3(512),  dim3(256), 0, stream>>>(Wv, wvh);
  k_f2bfT<<<dim3(512),  dim3(256), 0, stream>>>(Wp, wph);
  k_zero <<<dim3(64),   dim3(256), 0, stream>>>(mnv, (int)BT);

  dim3 gg(C_ / 128, (unsigned)(BT / 256));   // (8, 64)
  k_gemm<true> <<<gg, dim3(256), 0, stream>>>(xh, wqh, bq, qh, (int)BT, C_, C_);
  k_gemm<true> <<<gg, dim3(256), 0, stream>>>(xh, wkh, bk, kh, (int)BT, C_, C_);
  k_gemm<true> <<<gg, dim3(256), 0, stream>>>(xh, wvh, bv, vh, (int)BT, C_, C_);

  k_corr<<<dim3(T_ / 128, T_ / 256, B_), dim3(256), 0, stream>>>(qh, kh, mnv);
  k_topk<<<dim3(B_), dim3(256), 0, stream>>>(mnv, topw, topd);
  k_agg <<<dim3((T_ * C_) / 256, B_), dim3(256), 0, stream>>>(vh, topw, topd, aggh);

  k_gemm<false><<<gg, dim3(256), 0, stream>>>(aggh, wph, bp, out, (int)BT, C_, C_);
}
